// SVDLlamaBlock_85529978733224
// MI455X (gfx1250) — matrix operand order repaired and checked by reading, not compiled, not run on hardware
//
#include <hip/hip_runtime.h>
#include <hip/hip_bf16.h>

// ---------------------------------------------------------------------------
// SVD-Llama block for MI455X (gfx1250, wave32, WMMA).
// All heavy matmuls: bf16 v_wmma_f32_16x16x32_bf16. Attention is flash-style
// (no materialized [B,H,T,T] scores). Roofline: ~170 GFLOP vs ~300MB traffic
// => compute bound => WMMA everywhere; big GEMMs use 128x128 double-buffered
// LDS tiles (8 wmma / wave / K-step) fed by async global->LDS loads
// (ASYNCcnt path) when the toolchain exposes them.
// ---------------------------------------------------------------------------

typedef __attribute__((ext_vector_type(16))) __bf16 v16bf;
typedef __attribute__((ext_vector_type(8)))  float  v8f;

union V16BF { v16bf v; uint4 q[2]; };
static_assert(sizeof(V16BF) == 32, "V16BF size");

constexpr int B_   = 2;
constexpr int T_   = 1024;
constexpr int D_   = 2048;
constexpr int H_   = 32;
constexpr int HKV_ = 8;
constexpr int DH_  = 64;
constexpr int BT_  = B_ * T_;
constexpr float NEGBIG = -1e30f;

// Async global->LDS (CDNA5 ASYNCcnt path), guarded so absence of the builtin
// falls back to reg-staged copies and still compiles.
#if defined(__HIP_DEVICE_COMPILE__) && __has_builtin(__builtin_amdgcn_global_load_async_to_lds_b128)
#define ASYNC_LDS 1
#else
#define ASYNC_LDS 0
#endif

// The builtin takes typed 16-byte-vector pointers:
//   (v4i addrspace(1)* src, v4i addrspace(3)* dst, i32 offset, i32 cpol)
typedef int v4i __attribute__((vector_size(16)));
typedef v4i __attribute__((address_space(1)))* g_v4i_ptr;
typedef v4i __attribute__((address_space(3)))* l_v4i_ptr;

__device__ __forceinline__ g_v4i_ptr as_global_v4i(const void* p) {
    return (g_v4i_ptr)(unsigned long long)(size_t)p;
}
__device__ __forceinline__ l_v4i_ptr as_lds_v4i(void* p) {
    return (l_v4i_ptr)(unsigned)(size_t)p;
}

// ---------------------------------------------------------------------------
// Small batched WMMA GEMM:  C[M,N] = A[M,K](bf16,row) * B[N,K](bf16,row)^T
// Block tile 64x64, K-step 32, 256 threads = 8 waves (4 row-chunks x 2 col).
// Used for the tiny per-head Us projections (N=64, K=32, batched over heads).
// ---------------------------------------------------------------------------
template <typename OT>
__global__ __launch_bounds__(256)
void gemm_bf16_kernel(const __bf16* __restrict__ A, const __bf16* __restrict__ Bm,
                      OT* __restrict__ C,
                      int M, int N, int K, int lda, int ldb, int ldc,
                      long long sA, long long sB, long long sC)
{
    __shared__ __align__(16) __bf16 As[64][32];
    __shared__ __align__(16) __bf16 Bs[64][32];

    const int bz = blockIdx.z;
    A  += (long long)bz * sA;
    Bm += (long long)bz * sB;
    C  += (long long)bz * sC;

    const int tid  = threadIdx.x;
    const int wave = tid >> 5;
    const int lane = tid & 31;
    const int lrow = lane & 15;
    const int hi   = lane >> 4;
    const int wm   = wave >> 1;   // 0..3 : 16-row chunk
    const int wn   = wave & 1;    // 0..1 : 32-col chunk

    const int rowBase = blockIdx.y * 64;
    const int colBase = blockIdx.x * 64;

    const int ldr  = tid >> 2;        // 0..63
    const int ldc8 = (tid & 3) * 8;   // 0,8,16,24

    v8f acc0 = {};
    v8f acc1 = {};

    for (int k0 = 0; k0 < K; k0 += 32) {
        __syncthreads();
        {
            const __bf16* sa = A  + (long long)(rowBase + ldr) * lda + k0 + ldc8;
            *(uint4*)&As[ldr][ldc8] = *(const uint4*)sa;
            const __bf16* sb = Bm + (long long)(colBase + ldr) * ldb + k0 + ldc8;
            *(uint4*)&Bs[ldr][ldc8] = *(const uint4*)sb;
        }
        __syncthreads();

        V16BF a;
        a.q[0] = *(const uint4*)&As[wm * 16 + lrow][hi * 8];
        a.q[1] = *(const uint4*)&As[wm * 16 + lrow][16 + hi * 8];

        V16BF b0, b1;
        b0.q[0] = *(const uint4*)&Bs[wn * 32 + lrow][hi * 8];
        b0.q[1] = *(const uint4*)&Bs[wn * 32 + lrow][16 + hi * 8];
        b1.q[0] = *(const uint4*)&Bs[wn * 32 + 16 + lrow][hi * 8];
        b1.q[1] = *(const uint4*)&Bs[wn * 32 + 16 + lrow][16 + hi * 8];

        acc0 = __builtin_amdgcn_wmma_f32_16x16x32_bf16(false, a.v, false, b0.v,
                                                       (short)0, acc0, false, false);
        acc1 = __builtin_amdgcn_wmma_f32_16x16x32_bf16(false, a.v, false, b1.v,
                                                       (short)0, acc1, false, false);
    }

    const int c0 = colBase + wn * 32 + lrow;
#pragma unroll
    for (int v = 0; v < 8; ++v) {
        const int r = rowBase + wm * 16 + v + 8 * hi;
        C[(long long)r * ldc + c0]      = (OT)acc0[v];
        C[(long long)r * ldc + c0 + 16] = (OT)acc1[v];
    }
}

// ---------------------------------------------------------------------------
// Big WMMA GEMM: 128x128 block tile, K-step 32, double-buffered LDS.
// 256 threads = 8 waves (4 row x 2 col); each wave owns a 32x64 output
// (2 A-frags x 4 B-frags = 8 wmma per K-step). Tile staging for step i+1 is
// issued (async global->LDS when available) before the wmma burst of step i.
// Requires: M % 128 == 0, N % 128 == 0, K % 32 == 0.
// ---------------------------------------------------------------------------
template <typename OT>
__global__ __launch_bounds__(256)
void gemm_bf16_big_kernel(const __bf16* __restrict__ A, const __bf16* __restrict__ Bm,
                          OT* __restrict__ C, int K, int lda, int ldb, int ldc)
{
    __shared__ __align__(16) __bf16 As[2][128][32];
    __shared__ __align__(16) __bf16 Bs[2][128][32];

    const int tid  = threadIdx.x;
    const int wave = tid >> 5;
    const int lane = tid & 31;
    const int lrow = lane & 15;
    const int hi   = lane >> 4;
    const int wm   = wave & 3;    // 0..3 : 32-row chunk
    const int wn   = wave >> 2;   // 0..1 : 64-col chunk

    const int rowBase = blockIdx.y * 128;
    const int colBase = blockIdx.x * 128;

    // cooperative staging: 2 x 16B chunks per tile per thread (128x32 bf16)
    const int r0  = tid >> 1;              // rows 0..127
    const int o0  = (tid & 1) * 16;        // 0 or 16
    const int o0b = o0 + 8;

    v8f acc[2][4] = {};

    const __bf16* Arow = A  + (long long)(rowBase + r0) * lda;
    const __bf16* Brow = Bm + (long long)(colBase + r0) * ldb;

#if ASYNC_LDS
    auto issue_async = [&](int k0, int buf) {
        __builtin_amdgcn_global_load_async_to_lds_b128(
            as_global_v4i(Arow + k0 + o0),  as_lds_v4i(&As[buf][r0][o0]),  0, 0);
        __builtin_amdgcn_global_load_async_to_lds_b128(
            as_global_v4i(Arow + k0 + o0b), as_lds_v4i(&As[buf][r0][o0b]), 0, 0);
        __builtin_amdgcn_global_load_async_to_lds_b128(
            as_global_v4i(Brow + k0 + o0),  as_lds_v4i(&Bs[buf][r0][o0]),  0, 0);
        __builtin_amdgcn_global_load_async_to_lds_b128(
            as_global_v4i(Brow + k0 + o0b), as_lds_v4i(&Bs[buf][r0][o0b]), 0, 0);
    };
    issue_async(0, 0);
#else
    uint4 ra0, ra1, rb0, rb1;
    auto gload = [&](int k0) {
        ra0 = *(const uint4*)(Arow + k0 + o0);
        ra1 = *(const uint4*)(Arow + k0 + o0b);
        rb0 = *(const uint4*)(Brow + k0 + o0);
        rb1 = *(const uint4*)(Brow + k0 + o0b);
    };
    auto lstore = [&](int buf) {
        *(uint4*)&As[buf][r0][o0]  = ra0;
        *(uint4*)&As[buf][r0][o0b] = ra1;
        *(uint4*)&Bs[buf][r0][o0]  = rb0;
        *(uint4*)&Bs[buf][r0][o0b] = rb1;
    };
    gload(0);
    lstore(0);
#endif

    const int steps = K / 32;
    for (int it = 0; it < steps; ++it) {
        const int buf = it & 1;
        const bool more = (it + 1) < steps;

#if ASYNC_LDS
        // own async loads landed -> barrier -> everyone's landed
        asm volatile("s_wait_asynccnt 0" ::: "memory");
        __syncthreads();
        if (more) issue_async((it + 1) * 32, buf ^ 1);
#else
        __syncthreads();
        if (more) {
            gload((it + 1) * 32);
            if (it + 2 < steps) {
                __builtin_prefetch(Arow + (it + 2) * 32 + o0, 0, 1);
                __builtin_prefetch(Brow + (it + 2) * 32 + o0, 0, 1);
            }
        }
#endif

        // A fragments: lane row = lrow; VGPR0-3: K=hi*8+0..7, VGPR4-7: K=16+hi*8+0..7
        V16BF af[2];
#pragma unroll
        for (int a = 0; a < 2; ++a) {
            af[a].q[0] = *(const uint4*)&As[buf][wm * 32 + a * 16 + lrow][hi * 8];
            af[a].q[1] = *(const uint4*)&As[buf][wm * 32 + a * 16 + lrow][16 + hi * 8];
        }
        // B fragments: lane col = lrow; VGPR0-3: K=hi*8+0..7, VGPR4-7: K=16+hi*8+0..7
        V16BF bfr[4];
#pragma unroll
        for (int n = 0; n < 4; ++n) {
            bfr[n].q[0] = *(const uint4*)&Bs[buf][wn * 64 + n * 16 + lrow][hi * 8];
            bfr[n].q[1] = *(const uint4*)&Bs[buf][wn * 64 + n * 16 + lrow][16 + hi * 8];
        }

#pragma unroll
        for (int a = 0; a < 2; ++a)
#pragma unroll
            for (int n = 0; n < 4; ++n)
                acc[a][n] = __builtin_amdgcn_wmma_f32_16x16x32_bf16(
                    false, af[a].v, false, bfr[n].v, (short)0, acc[a][n], false, false);

#if !ASYNC_LDS
        if (more) lstore(buf ^ 1);
#endif
    }

    // C/D layout: VGPR v -> row v + 8*hi, col = lane%16
#pragma unroll
    for (int a = 0; a < 2; ++a) {
        const int rb = rowBase + wm * 32 + a * 16 + 8 * hi;
#pragma unroll
        for (int n = 0; n < 4; ++n) {
            const int cc = colBase + wn * 64 + n * 16 + lrow;
#pragma unroll
            for (int v = 0; v < 8; ++v)
                C[(long long)(rb + v) * ldc + cc] = (OT)acc[a][n][v];
        }
    }
}

// ---------------------------------------------------------------------------
// RMSNorm -> bf16
// ---------------------------------------------------------------------------
__global__ __launch_bounds__(256)
void rmsnorm_bf16_kernel(const float* __restrict__ x, const float* __restrict__ w,
                         __bf16* __restrict__ out, int D)
{
    const long long row = blockIdx.x;
    const float* xr = x + row * D;
    float ss = 0.f;
    for (int i = threadIdx.x; i < D; i += 256) { float v = xr[i]; ss += v * v; }
    __shared__ float red[8];
#pragma unroll
    for (int o = 16; o; o >>= 1) ss += __shfl_xor(ss, o, 32);
    if ((threadIdx.x & 31) == 0) red[threadIdx.x >> 5] = ss;
    __syncthreads();
    if (threadIdx.x < 8) {
        float t = red[threadIdx.x];
#pragma unroll
        for (int o = 4; o; o >>= 1) t += __shfl_xor(t, o, 8);
        if (threadIdx.x == 0) red[0] = t;
    }
    __syncthreads();
    const float inv = rsqrtf(red[0] / (float)D + 1e-5f);
    for (int i = threadIdx.x; i < D; i += 256)
        out[row * D + i] = (__bf16)(xr[i] * inv * w[i]);
}

// h = a + y ; x2 = rms(h)*w  (fused residual + second norm)
__global__ __launch_bounds__(256)
void add_rms_kernel(const float* __restrict__ a, const float* __restrict__ y,
                    const float* __restrict__ w, float* __restrict__ hout,
                    __bf16* __restrict__ xout, int D)
{
    const long long row = blockIdx.x;
    float ss = 0.f;
    for (int i = threadIdx.x; i < D; i += 256) {
        float t = a[row * D + i] + y[row * D + i];
        hout[row * D + i] = t;
        ss += t * t;
    }
    __shared__ float red[8];
#pragma unroll
    for (int o = 16; o; o >>= 1) ss += __shfl_xor(ss, o, 32);
    if ((threadIdx.x & 31) == 0) red[threadIdx.x >> 5] = ss;
    __syncthreads();
    if (threadIdx.x < 8) {
        float t = red[threadIdx.x];
#pragma unroll
        for (int o = 4; o; o >>= 1) t += __shfl_xor(t, o, 8);
        if (threadIdx.x == 0) red[0] = t;
    }
    __syncthreads();
    const float inv = rsqrtf(red[0] / (float)D + 1e-5f);
    for (int i = threadIdx.x; i < D; i += 256)
        xout[row * D + i] = (__bf16)(hout[row * D + i] * inv * w[i]);
}

// ---------------------------------------------------------------------------
// Elementwise helpers
// ---------------------------------------------------------------------------
__global__ void cast_f32_bf16_kernel(const float* __restrict__ src,
                                     __bf16* __restrict__ dst, long long n)
{
    long long i = (long long)blockIdx.x * blockDim.x + threadIdx.x;
    const long long stride = (long long)gridDim.x * blockDim.x;
    for (; i < n; i += stride) dst[i] = (__bf16)src[i];
}

__global__ void silu_mul_kernel(const __bf16* __restrict__ g, const __bf16* __restrict__ u,
                                __bf16* __restrict__ sw, long long n)
{
    long long i = (long long)blockIdx.x * blockDim.x + threadIdx.x;
    const long long stride = (long long)gridDim.x * blockDim.x;
    for (; i < n; i += stride) {
        float gf = (float)g[i];
        float s  = gf / (1.f + __expf(-gf));
        sw[i] = (__bf16)(s * (float)u[i]);
    }
}

__global__ void add_out_kernel(const float* __restrict__ h, const float* __restrict__ d,
                               float* __restrict__ out, long long n)
{
    long long i = (long long)blockIdx.x * blockDim.x + threadIdx.x;
    const long long stride = (long long)gridDim.x * blockDim.x;
    for (; i < n; i += stride) out[i] = h[i] + d[i];
}

// ---------------------------------------------------------------------------
// RoPE: interleaved-repeat angle table, gathered by position_ids.
// ---------------------------------------------------------------------------
__global__ __launch_bounds__(256)
void rope_kernel(const __bf16* __restrict__ in, __bf16* __restrict__ out,
                 const int* __restrict__ pos_ids, int nh)
{
    const int tok = blockIdx.x;
    const float pos = (float)pos_ids[tok];
    const int npair = nh * 32;
    for (int i = threadIdx.x; i < npair; i += 256) {
        const int head = i >> 5;
        const int d    = i & 31;
        const float invA = __powf(10000.f, -(float)(d & ~1) / 64.f);
        const float invB = __powf(10000.f, -(float)((d + 32) & ~1) / 64.f);
        const long long base = (long long)tok * (nh * 64) + head * 64;
        const float x0 = (float)in[base + d];
        const float x1 = (float)in[base + d + 32];
        const float aA = pos * invA, aB = pos * invB;
        out[base + d]      = (__bf16)(x0 * __cosf(aA) - x1 * __sinf(aA));
        out[base + d + 32] = (__bf16)(x1 * __cosf(aB) + x0 * __sinf(aB));
    }
}

// V pack-transpose: vbuf [BT, HKV*DH] -> vt [B, HKV, DH, T]
__global__ void vpack_kernel(const __bf16* __restrict__ vbuf, __bf16* __restrict__ vt)
{
    long long i = (long long)blockIdx.x * blockDim.x + threadIdx.x;
    const long long n = (long long)B_ * HKV_ * DH_ * T_;
    if (i >= n) return;
    const int t = (int)(i % T_);  long long r = i / T_;
    const int d = (int)(r % DH_); r /= DH_;
    const int h = (int)(r % HKV_);
    const int b = (int)(r / HKV_);
    vt[i] = vbuf[(long long)(b * T_ + t) * (HKV_ * DH_) + h * DH_ + d];
}

// ---------------------------------------------------------------------------
// Flash attention: one wave per 16-row Q tile, 32-key KV blocks.
//   S = Q*K^T : two chained wmma (DH=64 = 2 x K32) per 16-key score tile
//   P*V      : 4 wmma against pre-transposed V (B-fragments as two 16B row loads)
// Online softmax with shfl-width-16 row reductions on the C-layout halves.
// ---------------------------------------------------------------------------
__global__ __launch_bounds__(128)
void flash_attn_kernel(const __bf16* __restrict__ q,   // [BT, H*DH]
                       const __bf16* __restrict__ k,   // [BT, HKV*DH]
                       const __bf16* __restrict__ vt,  // [B, HKV, DH, T]
                       const int* __restrict__ amask,  // [B, T]
                       __bf16* __restrict__ ctx)       // [BT, H*DH]
{
    __shared__ __align__(16) __bf16 plds[4][16 * 32];

    const int wave = threadIdx.x >> 5;
    const int lane = threadIdx.x & 31;
    const int lrow = lane & 15;
    const int hi   = lane >> 4;

    const int bh  = blockIdx.y;       // b*H + h
    const int b   = bh >> 5;
    const int h   = bh & 31;
    const int kvh = h >> 2;           // GQA repeat = 4
    const int qtile = blockIdx.x * 4 + wave;
    const int qrow0 = qtile * 16;

    V16BF aq[2];
    {
        const __bf16* qp = q + (long long)(b * T_ + qrow0 + lrow) * (H_ * DH_) + h * DH_;
#pragma unroll
        for (int d = 0; d < 2; ++d) {
            aq[d].q[0] = *(const uint4*)(qp + d * 32 + hi * 8);
            aq[d].q[1] = *(const uint4*)(qp + d * 32 + 16 + hi * 8);
        }
    }

    v8f acc[4] = {{}, {}, {}, {}};
    float m[8], l[8];
#pragma unroll
    for (int v = 0; v < 8; ++v) { m[v] = NEGBIG; l[v] = 0.f; }

    const float scl = 0.125f;  // 1/sqrt(64)
    const int nkb = (qrow0 + 47) / 32;

    for (int kb = 0; kb < nkb; ++kb) {
        const int kbase = kb * 32;
        v8f s0 = {}, s1 = {};
#pragma unroll
        for (int d = 0; d < 2; ++d) {
            const __bf16* kp0 = k + (long long)(b * T_ + kbase + lrow) * (HKV_ * DH_)
                                  + kvh * DH_ + d * 32 + hi * 8;
            V16BF bk0; bk0.q[0] = *(const uint4*)kp0; bk0.q[1] = *(const uint4*)(kp0 + 16);
            s0 = __builtin_amdgcn_wmma_f32_16x16x32_bf16(false, aq[d].v, false, bk0.v,
                                                         (short)0, s0, false, false);
            const __bf16* kp1 = kp0 + (long long)16 * (HKV_ * DH_);
            V16BF bk1; bk1.q[0] = *(const uint4*)kp1; bk1.q[1] = *(const uint4*)(kp1 + 16);
            s1 = __builtin_amdgcn_wmma_f32_16x16x32_bf16(false, aq[d].v, false, bk1.v,
                                                         (short)0, s1, false, false);
        }

        const int kc0 = kbase + lrow;
        const int kc1 = kc0 + 16;
        const bool am0 = amask[b * T_ + kc0] != 0;
        const bool am1 = amask[b * T_ + kc1] != 0;
#pragma unroll
        for (int v = 0; v < 8; ++v) {
            const int qr = qrow0 + v + 8 * hi;
            float x0 = s0[v] * scl;
            float x1 = s1[v] * scl;
            if (kc0 > qr || !am0) x0 = NEGBIG;
            if (kc1 > qr || !am1) x1 = NEGBIG;
            float mx = fmaxf(x0, x1);
#pragma unroll
            for (int o = 8; o; o >>= 1) mx = fmaxf(mx, __shfl_xor(mx, o, 16));
            const float mn   = fmaxf(m[v], mx);
            const float corr = __expf(m[v] - mn);
            const float p0 = __expf(x0 - mn);
            const float p1 = __expf(x1 - mn);
            float rs = p0 + p1;
#pragma unroll
            for (int o = 8; o; o >>= 1) rs += __shfl_xor(rs, o, 16);
            l[v] = l[v] * corr + rs;
            m[v] = mn;
#pragma unroll
            for (int n = 0; n < 4; ++n) acc[n][v] *= corr;
            s0[v] = p0;
            s1[v] = p1;
        }

        // P (C-layout f32) -> LDS bf16 -> A-layout fragment (per-wave scratch;
        // same-wave DS ops are in-order; explicit counter wait blocks reorder)
        __bf16* pw = plds[wave];
#pragma unroll
        for (int v = 0; v < 8; ++v) {
            const int row = v + 8 * hi;
            pw[row * 32 + lrow]      = (__bf16)s0[v];
            pw[row * 32 + 16 + lrow] = (__bf16)s1[v];
        }
        asm volatile("s_wait_dscnt 0" ::: "memory");
        V16BF ap;
        ap.q[0] = *(const uint4*)&pw[lrow * 32 + hi * 8];
        ap.q[1] = *(const uint4*)&pw[lrow * 32 + 16 + hi * 8];

#pragma unroll
        for (int n = 0; n < 4; ++n) {
            const __bf16* vp = vt + ((long long)(b * HKV_ + kvh) * DH_ + n * 16 + lrow) * T_
                                  + kbase + hi * 8;
            V16BF bv; bv.q[0] = *(const uint4*)vp; bv.q[1] = *(const uint4*)(vp + 16);
            acc[n] = __builtin_amdgcn_wmma_f32_16x16x32_bf16(false, ap.v, false, bv.v,
                                                             (short)0, acc[n], false, false);
        }
    }

#pragma unroll
    for (int v = 0; v < 8; ++v) {
        const int qr = qrow0 + v + 8 * hi;
        const float inv = 1.0f / l[v];
        __bf16* op = ctx + (long long)(b * T_ + qr) * (H_ * DH_) + h * DH_ + lrow;
#pragma unroll
        for (int n = 0; n < 4; ++n) op[n * 16] = (__bf16)(acc[n][v] * inv);
    }
}

// ---------------------------------------------------------------------------
// Host-side orchestration
// ---------------------------------------------------------------------------
extern "C" void kernel_launch(void* const* d_in, const int* in_sizes, int n_in,
                              void* d_out, int out_size, void* d_ws, size_t ws_size,
                              hipStream_t stream)
{
    const float* hidden = (const float*)d_in[0];
    const float* ln1    = (const float*)d_in[1];
    const float* ln2    = (const float*)d_in[2];
    const float* qUs    = (const float*)d_in[3];
    const float* qV     = (const float*)d_in[4];
    const float* kUs    = (const float*)d_in[5];
    const float* kV     = (const float*)d_in[6];
    const float* vUs    = (const float*)d_in[7];
    const float* vV     = (const float*)d_in[8];
    const float* oUs    = (const float*)d_in[9];
    const float* oV     = (const float*)d_in[10];
    const float* gUs    = (const float*)d_in[11];
    const float* gV     = (const float*)d_in[12];
    const float* uUs    = (const float*)d_in[13];
    const float* uV     = (const float*)d_in[14];
    const float* dUs    = (const float*)d_in[15];
    const float* dV     = (const float*)d_in[16];
    const int*   amask  = (const int*)d_in[17];
    const int*   pos    = (const int*)d_in[18];

    char* wsp = (char*)d_ws;
    size_t off = 0;
    auto alloc = [&](size_t bytes) -> void* {
        void* p = wsp + off;
        off = (off + bytes + 255) & ~(size_t)255;
        return p;
    };

    // bf16 weight mirrors
    __bf16* bQV  = (__bf16*)alloc((size_t)1024 * 2048 * 2);
    __bf16* bQUs = (__bf16*)alloc((size_t)32 * 64 * 32 * 2);
    __bf16* bKV  = (__bf16*)alloc((size_t)256 * 2048 * 2);
    __bf16* bKUs = (__bf16*)alloc((size_t)8 * 64 * 32 * 2);
    __bf16* bVV  = (__bf16*)alloc((size_t)256 * 2048 * 2);
    __bf16* bVUs = (__bf16*)alloc((size_t)8 * 64 * 32 * 2);
    __bf16* bOUs = (__bf16*)alloc((size_t)2048 * 512 * 2);
    __bf16* bOV  = (__bf16*)alloc((size_t)512 * 2048 * 2);
    __bf16* bGUs = (__bf16*)alloc((size_t)8192 * 1024 * 2);
    __bf16* bGV  = (__bf16*)alloc((size_t)1024 * 2048 * 2);
    __bf16* bUUs = (__bf16*)alloc((size_t)8192 * 1024 * 2);
    __bf16* bUV  = (__bf16*)alloc((size_t)1024 * 2048 * 2);
    __bf16* bDUs = (__bf16*)alloc((size_t)2048 * 1024 * 2);
    __bf16* bDV  = (__bf16*)alloc((size_t)1024 * 8192 * 2);

    // activations
    __bf16* bX   = (__bf16*)alloc((size_t)BT_ * D_ * 2);
    __bf16* bXrq = (__bf16*)alloc((size_t)BT_ * 1024 * 2);
    __bf16* bXrk = (__bf16*)alloc((size_t)BT_ * 256 * 2);
    __bf16* bXrv = (__bf16*)alloc((size_t)BT_ * 256 * 2);
    __bf16* bQ   = (__bf16*)alloc((size_t)BT_ * H_ * DH_ * 2);
    __bf16* bK   = (__bf16*)alloc((size_t)BT_ * HKV_ * DH_ * 2);
    __bf16* bV   = (__bf16*)alloc((size_t)BT_ * HKV_ * DH_ * 2);
    __bf16* bQr  = (__bf16*)alloc((size_t)BT_ * H_ * DH_ * 2);
    __bf16* bKr  = (__bf16*)alloc((size_t)BT_ * HKV_ * DH_ * 2);
    __bf16* bVt  = (__bf16*)alloc((size_t)BT_ * HKV_ * DH_ * 2);
    __bf16* bCtx = (__bf16*)alloc((size_t)BT_ * H_ * DH_ * 2);
    __bf16* bT1  = (__bf16*)alloc((size_t)BT_ * 512 * 2);
    float*  fY   = (float*) alloc((size_t)BT_ * D_ * 4);
    float*  fH   = (float*) alloc((size_t)BT_ * D_ * 4);
    __bf16* bX2  = (__bf16*)alloc((size_t)BT_ * D_ * 2);
    __bf16* bGr  = (__bf16*)alloc((size_t)BT_ * 1024 * 2);
    __bf16* bUr  = (__bf16*)alloc((size_t)BT_ * 1024 * 2);
    __bf16* bG   = (__bf16*)alloc((size_t)BT_ * 8192 * 2);
    __bf16* bU   = (__bf16*)alloc((size_t)BT_ * 8192 * 2);
    __bf16* bSw  = (__bf16*)alloc((size_t)BT_ * 8192 * 2);
    __bf16* bDr  = (__bf16*)alloc((size_t)BT_ * 1024 * 2);
    float*  fD   = (float*) alloc((size_t)BT_ * D_ * 4);

    auto cast = [&](const float* s, __bf16* d, long long n) {
        cast_f32_bf16_kernel<<<2048, 256, 0, stream>>>(s, d, n);
    };
    auto gemmBigB = [&](const __bf16* A, const __bf16* Bm, __bf16* C, int M, int N, int K,
                        int lda, int ldb, int ldc) {
        dim3 g(N / 128, M / 128, 1);
        gemm_bf16_big_kernel<__bf16><<<g, 256, 0, stream>>>(A, Bm, C, K, lda, ldb, ldc);
    };
    auto gemmBigF = [&](const __bf16* A, const __bf16* Bm, float* C, int M, int N, int K,
                        int lda, int ldb, int ldc) {
        dim3 g(N / 128, M / 128, 1);
        gemm_bf16_big_kernel<float><<<g, 256, 0, stream>>>(A, Bm, C, K, lda, ldb, ldc);
    };
    auto gemmSmallB = [&](const __bf16* A, const __bf16* Bm, __bf16* C, int M, int N, int K,
                          int lda, int ldb, int ldc, int batch, long long sA, long long sB,
                          long long sC) {
        dim3 g(N / 64, M / 64, batch);
        gemm_bf16_kernel<__bf16><<<g, 256, 0, stream>>>(A, Bm, C, M, N, K, lda, ldb, ldc,
                                                        sA, sB, sC);
    };

    // --- weights -> bf16 ---
    cast(qV, bQV, 1024LL * 2048); cast(qUs, bQUs, 32LL * 64 * 32);
    cast(kV, bKV, 256LL * 2048);  cast(kUs, bKUs, 8LL * 64 * 32);
    cast(vV, bVV, 256LL * 2048);  cast(vUs, bVUs, 8LL * 64 * 32);
    cast(oUs, bOUs, 2048LL * 512); cast(oV, bOV, 512LL * 2048);
    cast(gUs, bGUs, 8192LL * 1024); cast(gV, bGV, 1024LL * 2048);
    cast(uUs, bUUs, 8192LL * 1024); cast(uV, bUV, 1024LL * 2048);
    cast(dUs, bDUs, 2048LL * 1024); cast(dV, bDV, 1024LL * 8192);

    // --- attention branch ---
    rmsnorm_bf16_kernel<<<BT_, 256, 0, stream>>>(hidden, ln1, bX, D_);

    gemmBigB(bX, bQV, bXrq, BT_, 1024, 2048, 2048, 2048, 1024);
    gemmBigB(bX, bKV, bXrk, BT_, 256, 2048, 2048, 2048, 256);
    gemmBigB(bX, bVV, bXrv, BT_, 256, 2048, 2048, 2048, 256);

    // per-head Us projections (batched over heads)
    gemmSmallB(bXrq, bQUs, bQ, BT_, 64, 32, 1024, 32, H_ * DH_, H_, 32, 64 * 32, 64);
    gemmSmallB(bXrk, bKUs, bK, BT_, 64, 32, 256, 32, HKV_ * DH_, HKV_, 32, 64 * 32, 64);
    gemmSmallB(bXrv, bVUs, bV, BT_, 64, 32, 256, 32, HKV_ * DH_, HKV_, 32, 64 * 32, 64);

    rope_kernel<<<BT_, 256, 0, stream>>>(bQ, bQr, pos, H_);
    rope_kernel<<<BT_, 256, 0, stream>>>(bK, bKr, pos, HKV_);
    vpack_kernel<<<(B_ * HKV_ * DH_ * T_) / 256, 256, 0, stream>>>(bV, bVt);

    flash_attn_kernel<<<dim3(T_ / 64, B_ * H_), 128, 0, stream>>>(bQr, bKr, bVt, amask, bCtx);

    gemmBigB(bCtx, bOV, bT1, BT_, 512, 2048, 2048, 2048, 512);
    gemmBigF(bT1, bOUs, fY, BT_, 2048, 512, 512, 512, 2048);

    add_rms_kernel<<<BT_, 256, 0, stream>>>(hidden, fY, ln2, fH, bX2, D_);

    // --- MLP branch ---
    gemmBigB(bX2, bGV, bGr, BT_, 1024, 2048, 2048, 2048, 1024);
    gemmBigB(bGr, bGUs, bG, BT_, 8192, 1024, 1024, 1024, 8192);
    gemmBigB(bX2, bUV, bUr, BT_, 1024, 2048, 2048, 2048, 1024);
    gemmBigB(bUr, bUUs, bU, BT_, 8192, 1024, 1024, 1024, 8192);

    silu_mul_kernel<<<4096, 256, 0, stream>>>(bG, bU, bSw, (long long)BT_ * 8192);

    gemmBigB(bSw, bDV, bDr, BT_, 1024, 8192, 8192, 8192, 1024);
    gemmBigF(bDr, bDUs, fD, BT_, 2048, 1024, 1024, 1024, 2048);

    add_out_kernel<<<2048, 256, 0, stream>>>(fH, fD, (float*)d_out, (long long)BT_ * D_);
}
